// MSDeformAttnTransformerEncoderLayer_77464030150788
// MI455X (gfx1250) — matrix operand order repaired and checked by reading, not compiled, not run on hardware
//
#include <hip/hip_runtime.h>
#include <hip/hip_bf16.h>

// ---------------------------------------------------------------------------
// MS-Deformable-Attention encoder layer for MI455X (gfx1250, wave32, WMMA).
// GEMMs run as bf16 WMMA (v_wmma_f32_16x16x32_bf16, f32 accumulate); A-tiles
// are staged to LDS with CDNA5 async copies (global_load_async_to_lds_b32,
// ASYNCcnt); bilinear sampling / softmax / layernorm stay fp32 VALU.
// ---------------------------------------------------------------------------

typedef float  v8f   __attribute__((ext_vector_type(8)));
typedef __bf16 v16bf __attribute__((ext_vector_type(16)));

__device__ __forceinline__ unsigned short f2bf(float f) {
    unsigned int u = __float_as_uint(f);
    u = (u + 0x7fffu + ((u >> 16) & 1u)) >> 16;   // round-to-nearest-even
    return (unsigned short)u;
}

union BFrag { v16bf v; unsigned int u[8]; };

// ---------------------------------------------------------------------------
// Generic GEMM:  C[M,N] = act(Abf[M,K] @ Wbf[K,N] + bias[N])
// 64x64 tile per 256-thread block (8 waves); each wave computes 16(M)x32(N)
// via two 16x16x32 bf16 WMMA per K-step. A staged via async mem->LDS copy,
// B staged manually (needs transpose into LDS).
// ---------------------------------------------------------------------------
template<bool RELU, bool OUT_BF>
__global__ __launch_bounds__(256) void gemm_bf16_wmma(
    const unsigned short* __restrict__ A,   // [M,K] bf16 row-major
    const unsigned short* __restrict__ W,   // [K,N] bf16 row-major
    const float* __restrict__ bias,         // [N]
    float* __restrict__ Cf,                 // out fp32 (if !OUT_BF)
    unsigned short* __restrict__ Cb,        // out bf16 (if OUT_BF)
    int K, int N)
{
    __shared__ unsigned short As[64][36];   // [m][k], padded rows
    __shared__ unsigned short Bs[64][36];   // [n][k] (transposed), padded

    const int tid  = threadIdx.x;
    const int lane = tid & 31, wid = tid >> 5;
    const int rm = (wid & 3) << 4;          // wave M offset inside tile
    const int cn = (wid >> 2) << 5;         // wave N offset inside tile
    const int lm = lane & 15, hi = lane >> 4;
    const int m0 = blockIdx.x << 6, n0 = blockIdx.y << 6;

    v8f acc0 = {0,0,0,0,0,0,0,0};
    v8f acc1 = {0,0,0,0,0,0,0,0};

    for (int k0 = 0; k0 < K; k0 += 32) {
        // ---- cooperative tile staging ----
#pragma unroll
        for (int i = 0; i < 4; ++i) {
            int u  = tid + (i << 8);
            // A: 64 rows x 32 k, direct memory->LDS async copy (ASYNCcnt),
            //    4B per lane, coalesced along k.
            int r  = u >> 4, kc = (u & 15) << 1;
            unsigned long long ga =
                (unsigned long long)(A + (size_t)(m0 + r) * K + (k0 + kc));
            unsigned ldsa = (unsigned)(size_t)(&As[r][kc]);   // LDS byte offset
            asm volatile("global_load_async_to_lds_b32 %0, %1, off"
                         :: "v"(ldsa), "v"(ga) : "memory");
            // B: 32 k x 64 n, load coalesced u32 and transpose into LDS.
            int kk = u >> 5, nc = (u & 31) << 1;
            unsigned int bv =
                *(const unsigned int*)(W + (size_t)(k0 + kk) * N + (n0 + nc));
            Bs[nc][kk]     = (unsigned short)(bv & 0xffffu);
            Bs[nc + 1][kk] = (unsigned short)(bv >> 16);
            if (i == 0)   // speculative prefetch of next W k-slab
                __builtin_prefetch(W + (size_t)(k0 + 32 + kk) * N + (n0 + nc), 0, 1);
        }
        asm volatile("s_wait_asynccnt 0x0" ::: "memory");
        __syncthreads();

        // ---- build fragments in contiguous k order (same for A and B) ----
        // A 16x32 (16-bit): lane%16 = M, VGPRj -> K=2j,2j+1 (+16 for hi half)
        // B 32x16 (16-bit): lane%16 = N, VGPRj -> K=2j,2j+1 (+16 for hi half)
        BFrag a, b0, b1;
#pragma unroll
        for (int j = 0; j < 8; ++j) {
            int ka = (j << 1) + (hi << 4);
            a.u[j]  = *(const unsigned int*)(&As[rm + lm][ka]);
            int kb = (j << 1) + (hi << 4);
            b0.u[j] = *(const unsigned int*)(&Bs[cn + lm][kb]);
            b1.u[j] = *(const unsigned int*)(&Bs[cn + 16 + lm][kb]);
        }
        acc0 = __builtin_amdgcn_wmma_f32_16x16x32_bf16(
                   false, a.v, false, b0.v, (short)0, acc0, false, false);
        acc1 = __builtin_amdgcn_wmma_f32_16x16x32_bf16(
                   false, a.v, false, b1.v, (short)0, acc1, false, false);
        __syncthreads();
    }

    // ---- epilogue: C/D layout: VGPR r -> M = r + 8*hi, N = lane%16 ----
#pragma unroll
    for (int r = 0; r < 8; ++r) {
        int m = m0 + rm + r + (hi << 3);
#pragma unroll
        for (int t = 0; t < 2; ++t) {
            int n = n0 + cn + (t << 4) + lm;
            float v = (t ? acc1[r] : acc0[r]) + bias[n];
            if (RELU) v = fmaxf(v, 0.0f);
            size_t o = (size_t)m * N + n;
            if (OUT_BF) Cb[o] = f2bf(v);
            else        Cf[o] = v;
        }
    }
}

// ---------------------------------------------------------------------------
// Elementwise converters
// ---------------------------------------------------------------------------
__global__ __launch_bounds__(256) void cvt_src_q(
    const float* __restrict__ src, const float* __restrict__ pos,
    unsigned short* __restrict__ src_bf, unsigned short* __restrict__ q_bf)
{
    size_t i = (size_t)blockIdx.x * 256 + threadIdx.x;
    float s = src[i];
    src_bf[i] = f2bf(s);
    q_bf[i]   = f2bf(s + pos[i]);
}

__global__ __launch_bounds__(256) void cvt_bf(
    const float* __restrict__ in, unsigned short* __restrict__ out)
{
    size_t i = (size_t)blockIdx.x * 256 + threadIdx.x;
    out[i] = f2bf(in[i]);
}

// ---------------------------------------------------------------------------
// Softmax over contiguous groups of 16 (NL*NP) — one thread per group
// ---------------------------------------------------------------------------
__global__ __launch_bounds__(256) void softmax16(float* __restrict__ a, int ngroups)
{
    int g = blockIdx.x * 256 + threadIdx.x;
    if (g >= ngroups) return;
    float* p = a + (size_t)g * 16;
    float m = p[0];
#pragma unroll
    for (int i = 1; i < 16; ++i) m = fmaxf(m, p[i]);
    float e[16]; float s = 0.f;
#pragma unroll
    for (int i = 0; i < 16; ++i) { e[i] = __expf(p[i] - m); s += e[i]; }
    float inv = 1.0f / s;
#pragma unroll
    for (int i = 0; i < 16; ++i) p[i] = e[i] * inv;
}

// ---------------------------------------------------------------------------
// Deformable sampling + attention reduction.
// One wave32 per (row, head); lane = head channel (HD=32) -> every value
// gather is a fully coalesced 128B line. Outputs bf16 for the w_out GEMM.
// ---------------------------------------------------------------------------
__global__ __launch_bounds__(256) void msda_sample(
    const float* __restrict__ value,   // [M,256]
    const float* __restrict__ off,     // [M,256] = (h,l,p,2)
    const float* __restrict__ attn,    // [M,128] = (h,l,p) softmaxed
    const int*   __restrict__ shapes,  // [4][2] (H,W)
    const int*   __restrict__ lstart,  // [4]
    unsigned short* __restrict__ outBf,// [M,256]
    int LEN)
{
    int lane = threadIdx.x & 31;                    // channel d
    int wid  = (blockIdx.x * 256 + threadIdx.x) >> 5;
    int row  = wid >> 3, h = wid & 7;
    int b = row / LEN, q = row - b * LEN;

    int sls[4] = { lstart[0], lstart[1], lstart[2], lstart[3] };
    int lq = (q >= sls[3]) ? 3 : (q >= sls[2]) ? 2 : (q >= sls[1]) ? 1 : 0;
    int local = q - sls[lq];
    int Hq = shapes[lq * 2], Wq = shapes[lq * 2 + 1];
    float refx = ((float)(local % Wq) + 0.5f) / (float)Wq;
    float refy = ((float)(local / Wq) + 0.5f) / (float)Hq;

    const float* offp = off  + (size_t)row * 256 + h * 32;
    const float* attp = attn + (size_t)row * 128 + h * 16;

    float acc = 0.0f;
#pragma unroll
    for (int l = 0; l < 4; ++l) {
        int Hl = shapes[l * 2], Wl = shapes[l * 2 + 1];
        const float* vbase =
            value + ((size_t)b * LEN + sls[l]) * 256 + h * 32 + lane;
#pragma unroll
        for (int p = 0; p < 4; ++p) {
            float ox = offp[l * 8 + p * 2 + 0];
            float oy = offp[l * 8 + p * 2 + 1];
            float aw = attp[l * 4 + p];
            float x = (refx + ox / (float)Wl) * (float)Wl - 0.5f;
            float y = (refy + oy / (float)Hl) * (float)Hl - 0.5f;
            float x0f = floorf(x), y0f = floorf(y);
            float dx = x - x0f, dy = y - y0f;
            int x0 = (int)x0f, y0 = (int)y0f;
            float wt[4] = { (1.f - dx) * (1.f - dy), dx * (1.f - dy),
                            (1.f - dx) * dy,         dx * dy };
#pragma unroll
            for (int t = 0; t < 4; ++t) {
                int xi = x0 + (t & 1), yi = y0 + (t >> 1);
                bool valid = (xi >= 0) && (xi < Wl) && (yi >= 0) && (yi < Hl);
                int xc = min(max(xi, 0), Wl - 1);
                int yc = min(max(yi, 0), Hl - 1);
                float v = vbase[(size_t)(yc * Wl + xc) * 256];
                acc += aw * wt[t] * (valid ? v : 0.0f);
            }
        }
    }
    outBf[(size_t)row * 256 + h * 32 + lane] = f2bf(acc);
}

// ---------------------------------------------------------------------------
// LayerNorm with residual: y = LN(xin + res)*g + b. One wave32 per 256-row.
// ---------------------------------------------------------------------------
__global__ __launch_bounds__(256) void ln_residual(
    const float* __restrict__ xin, const float* __restrict__ res,
    const float* __restrict__ gam, const float* __restrict__ bet,
    float* __restrict__ outF, unsigned short* __restrict__ outB)
{
    int lane = threadIdx.x & 31;
    int row  = (blockIdx.x << 3) + (threadIdx.x >> 5);
    size_t base = (size_t)row * 256;

    float v[8]; float s = 0.f, sq = 0.f;
#pragma unroll
    for (int i = 0; i < 8; ++i) {
        int c = lane + (i << 5);
        float x = xin[base + c] + res[base + c];
        v[i] = x; s += x; sq += x * x;
    }
#pragma unroll
    for (int m = 16; m >= 1; m >>= 1) {
        s  += __shfl_xor(s,  m, 32);
        sq += __shfl_xor(sq, m, 32);
    }
    float mean = s * (1.0f / 256.0f);
    float var  = sq * (1.0f / 256.0f) - mean * mean;
    float rstd = rsqrtf(var + 1e-5f);
#pragma unroll
    for (int i = 0; i < 8; ++i) {
        int c = lane + (i << 5);
        float y = (v[i] - mean) * rstd * gam[c] + bet[c];
        if (outF) outF[base + c] = y;
        if (outB) outB[base + c] = f2bf(y);
    }
}

// ---------------------------------------------------------------------------
// Orchestration
// ---------------------------------------------------------------------------
extern "C" void kernel_launch(void* const* d_in, const int* in_sizes, int n_in,
                              void* d_out, int out_size, void* d_ws, size_t ws_size,
                              hipStream_t stream) {
    const float* src    = (const float*)d_in[0];
    const float* pos    = (const float*)d_in[1];
    const int*   shapes = (const int*)d_in[2];
    const int*   lstart = (const int*)d_in[3];
    const float* w_value= (const float*)d_in[4];
    const float* b_value= (const float*)d_in[5];
    const float* w_off  = (const float*)d_in[6];
    const float* b_off  = (const float*)d_in[7];
    const float* w_attn = (const float*)d_in[8];
    const float* b_attn = (const float*)d_in[9];
    const float* w_out  = (const float*)d_in[10];
    const float* b_out  = (const float*)d_in[11];
    const float* w1     = (const float*)d_in[12];
    const float* b1     = (const float*)d_in[13];
    const float* w2     = (const float*)d_in[14];
    const float* b2     = (const float*)d_in[15];
    const float* g1     = (const float*)d_in[16];
    const float* be1    = (const float*)d_in[17];
    const float* g2     = (const float*)d_in[18];
    const float* be2    = (const float*)d_in[19];

    const int D = 256, DFF = 2048;
    const int Mrows = in_sizes[0] / D;     // 10880 (multiple of 64)
    const int LEN   = Mrows / 2;           // B = 2

    // ---- workspace carve (256B aligned) ----
    char* wsp = (char*)d_ws;
    auto carve = [&](size_t bytes) -> void* {
        void* p = wsp; wsp += (bytes + 255) & ~(size_t)255; return p;
    };
    unsigned short* src_bf    = (unsigned short*)carve((size_t)Mrows * D * 2);
    unsigned short* q_bf      = (unsigned short*)carve((size_t)Mrows * D * 2);
    unsigned short* wv_bf     = (unsigned short*)carve((size_t)D * D * 2);
    unsigned short* woff_bf   = (unsigned short*)carve((size_t)D * D * 2);
    unsigned short* wattn_bf  = (unsigned short*)carve((size_t)D * 128 * 2);
    unsigned short* wout_bf   = (unsigned short*)carve((size_t)D * D * 2);
    unsigned short* w1_bf     = (unsigned short*)carve((size_t)D * DFF * 2);
    unsigned short* w2_bf     = (unsigned short*)carve((size_t)DFF * D * 2);
    float* value_f            = (float*)carve((size_t)Mrows * D * 4);
    float* off_f              = (float*)carve((size_t)Mrows * D * 4);
    float* attn_f             = (float*)carve((size_t)Mrows * 128 * 4);
    unsigned short* attnout_bf= (unsigned short*)carve((size_t)Mrows * D * 2);
    unsigned short* h_bf      = (unsigned short*)carve((size_t)Mrows * DFF * 2);
    // aliases (lifetimes do not overlap):
    float* gout_f = off_f;               // w_out GEMM output (off consumed)
    float* x_f    = value_f;             // LN1 fp32 output (value consumed)
    unsigned short* x_bf = attnout_bf;   // LN1 bf16 output (attnout consumed)
    float* y_f    = off_f;               // FFN2 output (gout consumed by LN1)

    // ---- P1: bf16 conversions ----
    cvt_src_q<<<Mrows, 256, 0, stream>>>(src, pos, src_bf, q_bf);
    cvt_bf<<<(D * D)   / 256, 256, 0, stream>>>(w_value, wv_bf);
    cvt_bf<<<(D * D)   / 256, 256, 0, stream>>>(w_off,   woff_bf);
    cvt_bf<<<(D * 128) / 256, 256, 0, stream>>>(w_attn,  wattn_bf);
    cvt_bf<<<(D * D)   / 256, 256, 0, stream>>>(w_out,   wout_bf);
    cvt_bf<<<(D * DFF) / 256, 256, 0, stream>>>(w1,      w1_bf);
    cvt_bf<<<(DFF * D) / 256, 256, 0, stream>>>(w2,      w2_bf);

    const int MB = Mrows / 64;   // 170
    // ---- P2: projection GEMMs ----
    gemm_bf16_wmma<false, false><<<dim3(MB, D / 64),   256, 0, stream>>>(
        src_bf, wv_bf, b_value, value_f, nullptr, D, D);
    gemm_bf16_wmma<false, false><<<dim3(MB, D / 64),   256, 0, stream>>>(
        q_bf, woff_bf, b_off, off_f, nullptr, D, D);
    gemm_bf16_wmma<false, false><<<dim3(MB, 128 / 64), 256, 0, stream>>>(
        q_bf, wattn_bf, b_attn, attn_f, nullptr, D, 128);

    // ---- P3: softmax over 16 (NL*NP) per (b,q,h) ----
    softmax16<<<(Mrows * 8) / 256, 256, 0, stream>>>(attn_f, Mrows * 8);

    // ---- P4: deformable sampling + attention reduce ----
    msda_sample<<<Mrows, 256, 0, stream>>>(value_f, off_f, attn_f,
                                           shapes, lstart, attnout_bf, LEN);

    // ---- P5: output projection ----
    gemm_bf16_wmma<false, false><<<dim3(MB, D / 64), 256, 0, stream>>>(
        attnout_bf, wout_bf, b_out, gout_f, nullptr, D, D);

    // ---- P6: LN1(src + attn_out) -> x (fp32 + bf16) ----
    ln_residual<<<Mrows / 8, 256, 0, stream>>>(gout_f, src, g1, be1, x_f, x_bf);

    // ---- P7: FFN1 with ReLU -> bf16 hidden ----
    gemm_bf16_wmma<true, true><<<dim3(MB, DFF / 64), 256, 0, stream>>>(
        x_bf, w1_bf, b1, nullptr, h_bf, D, DFF);

    // ---- P8: FFN2 ----
    gemm_bf16_wmma<false, false><<<dim3(MB, D / 64), 256, 0, stream>>>(
        h_bf, w2_bf, b2, y_f, nullptr, DFF, D);

    // ---- P9: LN2(x + ffn) -> d_out ----
    ln_residual<<<Mrows / 8, 256, 0, stream>>>(y_f, x_f, g2, be2,
                                               (float*)d_out, nullptr);
}
